// FeedbackTransformer_57363583205912
// MI455X (gfx1250) — matrix operand order repaired and checked by reading, not compiled, not run on hardware
//
#include <hip/hip_runtime.h>
#include <hip/hip_bf16.h>
#include <math.h>

// ---------------------------------------------------------------------------
// Model constants (from reference)
// ---------------------------------------------------------------------------
#define VOCAB     32000
#define DIM       512
#define DEPTH     6
#define SEQLEN    256
#define BATCH     8
#define SEQ_CHUNK 16
#define HEADS     8
#define DIM_HEAD  64
#define INNER     512            // HEADS * DIM_HEAD
#define NCHUNK    (SEQLEN / SEQ_CHUNK)     // 16
#define MTOK      (BATCH * SEQ_CHUNK)      // 128 tokens per chunk
#define FF_HID    2048           // DIM*4
#define FF_WIDE   4096           // DIM*8
#define LN_EPS    1e-5f

typedef __attribute__((ext_vector_type(2)))  __bf16 v2bf;
typedef __attribute__((ext_vector_type(4)))  __bf16 v4bf;
typedef __attribute__((ext_vector_type(8)))  __bf16 v8bf;
typedef __attribute__((ext_vector_type(16))) __bf16 v16bf;
typedef __attribute__((ext_vector_type(4)))  float  v4f;
typedef __attribute__((ext_vector_type(8)))  float  v8f;

// ---------------------------------------------------------------------------
// bf16 WMMA GEMM:  C[M,N] = alpha * A[M,K] @ B[K,N] (+ bias[N]) (+ res[M,N])
//
// Block = 128 threads (4 waves), tile = 16(M) x 128(N). Each wave owns two
// 16x16 output tiles sharing one A fragment -> 2x v_wmma_f32_16x16x32_bf16
// per wave per K-step. LDS layouts are chosen so fragment builds are wide:
//   sA  [m][k] row-major : A fragment = 2x ds_load_b128 per lane
//   sBT [n][k] transposed: B fragment = 2x ds_load_b128 per lane (K-contig)
// Requires: M%16==0, N%128==0, K%32==0 (all shapes in this model satisfy it).
// ---------------------------------------------------------------------------
__global__ __launch_bounds__(128)
void gemm_bf16_wmma(const float* __restrict__ A, const float* __restrict__ B,
                    float* __restrict__ C,
                    const float* __restrict__ bias,
                    const float* __restrict__ res,
                    int M, int N, int K, float alpha)
{
    __shared__ __attribute__((aligned(64))) __bf16 sA[16 * 32];    // [m][k]
    __shared__ __attribute__((aligned(64))) __bf16 sBT[128 * 32];  // [n][k]

    const int tid  = threadIdx.x;
    const int wv   = tid >> 5;
    const int lane = tid & 31;
    const int rowBase = blockIdx.y * 16;
    const int colBase = blockIdx.x * 128;

    v8f acc0 = {}, acc1 = {};

    // Static staging roles
    const int mA = tid >> 3;          // 0..15
    const int k4 = (tid & 7) << 2;    // 0,4,..,28

    for (int k0 = 0; k0 < K; k0 += 32) {
        // ---- branch-free prefetch of next K-tile (global_prefetch_b8).
        //      Clamp via select; re-prefetching the last tile is harmless. ----
        {
            const int kpf = (k0 + 32 < K) ? (k0 + 32) : k0;
            __builtin_prefetch(A + (size_t)(rowBase + mA) * K + kpf + k4, 0, 0);
            __builtin_prefetch(B + (size_t)(kpf + (tid & 31)) * N + colBase + (wv << 5), 0, 0);
        }

        // ---- stage A (16x32): one float4 load + one 8B LDS store / thread ----
        {
            v4f f = *(const v4f*)(A + (size_t)(rowBase + mA) * K + k0 + k4);
            v4bf t;
            t[0] = (__bf16)f[0]; t[1] = (__bf16)f[1];
            t[2] = (__bf16)f[2]; t[3] = (__bf16)f[3];
            *(v4bf*)(sA + mA * 32 + k4) = t;
        }
        // ---- stage B (32x128) transposed into [n][k], packing k-pairs ----
#pragma unroll
        for (int it = 0; it < 4; ++it) {
            const int idx = tid + (it << 7);
            const int n4  = (idx & 31) << 2;   // 0..124
            const int kp  = idx >> 5;          // 0..15 (k-pair)
            const float* bp = B + (size_t)(k0 + 2 * kp) * N + colBase + n4;
            v4f f0 = *(const v4f*)bp;
            v4f f1 = *(const v4f*)(bp + N);
#pragma unroll
            for (int jj = 0; jj < 4; ++jj) {
                v2bf t; t[0] = (__bf16)f0[jj]; t[1] = (__bf16)f1[jj];
                *(v2bf*)(sBT + (n4 + jj) * 32 + 2 * kp) = t;
            }
        }
        __syncthreads();

        // ---- fragments (contiguous k = 16*(lane/16) + s for A and B, wave32) ----
        // A 16x32: lane L -> row m=L%16; lanes<16 hold K 0..15,
        //          lanes>=16 hold K 16..31 (matches B fragment).
        const int m  = lane & 15;
        const int hi = (lane >> 4) << 4;         // 0 or 16
        union { v16bf v; v8bf h[2]; } ua;
        ua.h[0] = *(const v8bf*)(sA + m * 32 + hi);
        ua.h[1] = *(const v8bf*)(sA + m * 32 + hi + 8);

        // B 32x16: lane L -> col n=L%16; lanes<16 K 0..15, lanes>=16 K 16..31.
        const int kof  = (lane >> 4) << 4;       // 0 or 16
        const int nloc = (lane & 15) + (wv << 5);
        v16bf fb0 = *(const v16bf*)(sBT + nloc * 32 + kof);
        v16bf fb1 = *(const v16bf*)(sBT + (nloc + 16) * 32 + kof);

        acc0 = __builtin_amdgcn_wmma_f32_16x16x32_bf16(
                   false, ua.v, false, fb0, (short)0, acc0, false, false);
        acc1 = __builtin_amdgcn_wmma_f32_16x16x32_bf16(
                   false, ua.v, false, fb1, (short)0, acc1, false, false);
        __syncthreads();
    }

    // ---- epilogue (uniform branches). C/D layout: VGPR r -> M=r+8*(lane/16),
    //      N=lane%16. ----
    const int n0 = (lane & 15) + (wv << 5) + colBase;
    const int n1 = n0 + 16;
    const int mB = ((lane >> 4) << 3) + rowBase;
    const float bv0 = bias ? bias[n0] : 0.f;
    const float bv1 = bias ? bias[n1] : 0.f;
    if (res) {
#pragma unroll
        for (int r = 0; r < 8; ++r) {
            const size_t row = (size_t)(mB + r) * N;
            C[row + n0] = acc0[r] * alpha + bv0 + res[row + n0];
            C[row + n1] = acc1[r] * alpha + bv1 + res[row + n1];
        }
    } else {
#pragma unroll
        for (int r = 0; r < 8; ++r) {
            const size_t row = (size_t)(mB + r) * N;
            C[row + n0] = acc0[r] * alpha + bv0;
            C[row + n1] = acc1[r] * alpha + bv1;
        }
    }
}

// ---------------------------------------------------------------------------
// Token embedding gather: x[b,n,:] = tok_emb[tokens[b,n], :]
// ---------------------------------------------------------------------------
__global__ void embed_kernel(const int* __restrict__ tokens,
                             const float* __restrict__ tok_emb,
                             float* __restrict__ x)
{
    const int row = blockIdx.x;                 // 0..2047 (b*256+n)
    const int tk  = tokens[row];
    for (int d = threadIdx.x; d < DIM; d += blockDim.x)
        x[(size_t)row * DIM + d] = tok_emb[(size_t)tk * DIM + d];
}

// ---------------------------------------------------------------------------
// LayerNorm over last dim (DIM=512): one row per 128-thread block.
// ---------------------------------------------------------------------------
__global__ __launch_bounds__(128)
void ln_kernel(const float* __restrict__ x, const float* __restrict__ g,
               const float* __restrict__ b, float* __restrict__ y)
{
    const int row = blockIdx.x;
    const float* xr = x + (size_t)row * DIM;
    __shared__ float red[128];

    float v[4];
    float s = 0.f;
#pragma unroll
    for (int i = 0; i < 4; ++i) { v[i] = xr[threadIdx.x + i * 128]; s += v[i]; }
    red[threadIdx.x] = s; __syncthreads();
    for (int st = 64; st > 0; st >>= 1) {
        if (threadIdx.x < st) red[threadIdx.x] += red[threadIdx.x + st];
        __syncthreads();
    }
    const float mu = red[0] * (1.f / DIM);
    __syncthreads();

    float s2 = 0.f;
#pragma unroll
    for (int i = 0; i < 4; ++i) { const float d = v[i] - mu; s2 += d * d; }
    red[threadIdx.x] = s2; __syncthreads();
    for (int st = 64; st > 0; st >>= 1) {
        if (threadIdx.x < st) red[threadIdx.x] += red[threadIdx.x + st];
        __syncthreads();
    }
    const float rstd = rsqrtf(red[0] * (1.f / DIM) + LN_EPS);
#pragma unroll
    for (int i = 0; i < 4; ++i) {
        const int col = threadIdx.x + i * 128;
        y[(size_t)row * DIM + col] = (v[i] - mu) * rstd * g[col] + b[col];
    }
}

// ---------------------------------------------------------------------------
// Attention for one chunk. grid = BATCH*HEADS blocks, 256 threads.
// q:   [128, INNER]  (already scaled by 1/sqrt(d))
// kv:  [128, 2*INNER] (k = [:,0:512], v = [:,512:1024]) current chunk
// memk/memv: [BATCH, 256, INNER], first memLen keys valid.
// Rel-pos bucket reduces to max(qi-kj,0) (<=15 < NUM_BUCKETS/2).
// Causal mask on the trailing 16 (current) keys only. Mem/current key loops
// are split so the hot dot-product loops carry no per-iteration branch.
// ---------------------------------------------------------------------------
__global__ __launch_bounds__(256)
void attn_kernel(const float* __restrict__ q, const float* __restrict__ kv,
                 const float* __restrict__ memk, const float* __restrict__ memv,
                 int memLen, const float* __restrict__ relEmb,
                 float* __restrict__ out)
{
    const int b  = blockIdx.x >> 3;
    const int hh = blockIdx.x & 7;
    const int j  = memLen + SEQ_CHUNK;
    const int tid = threadIdx.x;

    __shared__ float qs[SEQ_CHUNK * DIM_HEAD];       // 4 KB
    __shared__ float sim[SEQ_CHUNK * 256];           // 16 KB
    __shared__ float rstat[SEQ_CHUNK * 16];
    __shared__ float rowmax[SEQ_CHUNK], rowsum[SEQ_CHUNK];

    for (int i = tid; i < SEQ_CHUNK * DIM_HEAD; i += 256) {
        const int qi = i >> 6, d = i & 63;
        qs[i] = q[(size_t)(b * SEQ_CHUNK + qi) * INNER + hh * DIM_HEAD + d];
    }
    __syncthreads();

    // scores vs memory keys
    for (int i = tid; i < SEQ_CHUNK * memLen; i += 256) {
        const int qi = i & 15, kj = i >> 4;
        const float* kp = memk + ((size_t)(b * SEQLEN + kj)) * INNER + hh * DIM_HEAD;
        float s = 0.f;
#pragma unroll 8
        for (int d = 0; d < DIM_HEAD; ++d) s += qs[qi * DIM_HEAD + d] * kp[d];
        const int bucket = (qi > kj) ? (qi - kj) : 0;
        sim[qi * 256 + kj] = s + relEmb[bucket * HEADS + hh];
    }
    // scores vs current-chunk keys (+causal mask)
    for (int i = tid; i < SEQ_CHUNK * SEQ_CHUNK; i += 256) {
        const int qi = i & 15, kc = i >> 4;      // kc = key idx within chunk
        const int kj = memLen + kc;
        const float* kp = kv + ((size_t)(b * SEQ_CHUNK + kc)) * (2 * INNER) + hh * DIM_HEAD;
        float s = 0.f;
#pragma unroll 8
        for (int d = 0; d < DIM_HEAD; ++d) s += qs[qi * DIM_HEAD + d] * kp[d];
        const int bucket = (qi > kj) ? (qi - kj) : 0;
        s += relEmb[bucket * HEADS + hh];
        if (kc > qi) s = -3.0e38f;
        sim[qi * 256 + kj] = s;
    }
    __syncthreads();

    // softmax (row = tid/16, 16 threads per row)
    const int rr = tid >> 4, sub = tid & 15;
    float mx = -3.4e38f;
    for (int kj = sub; kj < j; kj += 16) mx = fmaxf(mx, sim[rr * 256 + kj]);
    rstat[rr * 16 + sub] = mx; __syncthreads();
    if (sub == 0) {
        float m2 = -3.4e38f;
        for (int t = 0; t < 16; ++t) m2 = fmaxf(m2, rstat[rr * 16 + t]);
        rowmax[rr] = m2;
    }
    __syncthreads();
    float sm = 0.f;
    for (int kj = sub; kj < j; kj += 16) {
        const float e = __expf(sim[rr * 256 + kj] - rowmax[rr]);
        sim[rr * 256 + kj] = e;
        sm += e;
    }
    rstat[rr * 16 + sub] = sm; __syncthreads();
    if (sub == 0) {
        float t2 = 0.f;
        for (int t = 0; t < 16; ++t) t2 += rstat[rr * 16 + t];
        rowsum[rr] = t2;
    }
    __syncthreads();

    // out = attn @ v  (mem part + current part, branch-free inner loops)
    for (int i = tid; i < SEQ_CHUNK * DIM_HEAD; i += 256) {
        const int qi = i >> 6, d = i & 63;
        float acc = 0.f;
        for (int kj = 0; kj < memLen; ++kj)
            acc += sim[qi * 256 + kj] *
                   memv[((size_t)(b * SEQLEN + kj)) * INNER + hh * DIM_HEAD + d];
#pragma unroll
        for (int kc = 0; kc < SEQ_CHUNK; ++kc)
            acc += sim[qi * 256 + memLen + kc] *
                   kv[((size_t)(b * SEQ_CHUNK + kc)) * (2 * INNER) + INNER + hh * DIM_HEAD + d];
        out[(size_t)(b * SEQ_CHUNK + qi) * INNER + hh * DIM_HEAD + d] = acc / rowsum[qi];
    }
}

// ---------------------------------------------------------------------------
// GEGLU: ff = gelu_exact(u[:,2048:4096]) * u[:,0:2048]
// ---------------------------------------------------------------------------
__global__ void gelu_gate_kernel(const float* __restrict__ u, float* __restrict__ ff)
{
    const int i = blockIdx.x * blockDim.x + threadIdx.x;   // 128*2048 elems
    const int r = i >> 11, c = i & 2047;
    const float a  = u[(size_t)r * FF_WIDE + c];
    const float gt = u[(size_t)r * FF_WIDE + FF_HID + c];
    const float ge = 0.5f * gt * (1.f + erff(gt * 0.70710678118654752f));
    ff[(size_t)r * FF_HID + c] = ge * a;
}

// ---------------------------------------------------------------------------
// Small data-movement / aggregation kernels
// ---------------------------------------------------------------------------
__global__ void copy_f32(const float* __restrict__ s, float* __restrict__ d, int n)
{
    const int i = blockIdx.x * blockDim.x + threadIdx.x;
    if (i < n) d[i] = s[i];
}

__global__ void gather_chunk(const float* __restrict__ x, float* __restrict__ xc, int c)
{
    const int i = blockIdx.x * blockDim.x + threadIdx.x;   // 128*512
    const int r = i >> 9, d = i & 511;
    const int b = r >> 4, t = r & 15;
    xc[i] = x[(size_t)(b * SEQLEN + c * SEQ_CHUNK + t) * DIM + d];
}

__global__ void scatter_chunk(const float* __restrict__ xc, float* __restrict__ x, int c)
{
    const int i = blockIdx.x * blockDim.x + threadIdx.x;
    const int r = i >> 9, d = i & 511;
    const int b = r >> 4, t = r & 15;
    x[(size_t)(b * SEQLEN + c * SEQ_CHUNK + t) * DIM + d] = xc[i];
}

__global__ void agg_kernel(const float* __restrict__ hid, const float* __restrict__ lw,
                           float* __restrict__ agg)
{
    const int i = blockIdx.x * blockDim.x + threadIdx.x;   // 128*512
    float s = 0.f;
#pragma unroll
    for (int l = 0; l < DEPTH; ++l) s += lw[l] * hid[(size_t)l * (MTOK * DIM) + i];
    agg[i] = s;
}

__global__ void scatter_memkv(const float* __restrict__ mkv, float* __restrict__ memk,
                              float* __restrict__ memv, int c)
{
    const int i = blockIdx.x * blockDim.x + threadIdx.x;   // 128*512
    const int r = i >> 9, d = i & 511;
    const int b = r >> 4, t = r & 15;
    const size_t dst = (size_t)(b * SEQLEN + c * SEQ_CHUNK + t) * INNER + d;
    memk[dst] = mkv[(size_t)r * (2 * INNER) + d];
    memv[dst] = mkv[(size_t)r * (2 * INNER) + INNER + d];
}

// ---------------------------------------------------------------------------
// Host orchestration
// ---------------------------------------------------------------------------
extern "C" void kernel_launch(void* const* d_in, const int* in_sizes, int n_in,
                              void* d_out, int out_size, void* d_ws, size_t ws_size,
                              hipStream_t stream)
{
    (void)in_sizes; (void)n_in; (void)out_size; (void)ws_size;

    const int*   tokens   = (const int*)  d_in[0];
    const float* tok_emb  = (const float*)d_in[1];
    const float* rel_emb  = (const float*)d_in[2];
    const float* lw       = (const float*)d_in[3];
    const float* W_mem_kv = (const float*)d_in[4];
    const float* ln1_g    = (const float*)d_in[5];
    const float* ln1_b    = (const float*)d_in[6];
    const float* Wq       = (const float*)d_in[7];
    const float* Wkv      = (const float*)d_in[8];
    const float* Wo       = (const float*)d_in[9];
    const float* bo       = (const float*)d_in[10];
    const float* ln2_g    = (const float*)d_in[11];
    const float* ln2_b    = (const float*)d_in[12];
    const float* W1       = (const float*)d_in[13];
    const float* b1       = (const float*)d_in[14];
    const float* W2       = (const float*)d_in[15];
    const float* b2       = (const float*)d_in[16];
    const float* lnf_g    = (const float*)d_in[17];
    const float* lnf_b    = (const float*)d_in[18];
    const float* Wlog     = (const float*)d_in[19];
    const float* blog     = (const float*)d_in[20];
    float* logits = (float*)d_out;

    // Workspace layout (floats)
    float* ws = (float*)d_ws;
    size_t off = 0;
    float* x        = ws + off; off += (size_t)BATCH * SEQLEN * DIM;      // 1,048,576
    float* xn       = ws + off; off += (size_t)BATCH * SEQLEN * DIM;      // 1,048,576
    float* mem_k    = ws + off; off += (size_t)BATCH * SEQLEN * INNER;    // 1,048,576
    float* mem_v    = ws + off; off += (size_t)BATCH * SEQLEN * INNER;    // 1,048,576
    float* xc       = ws + off; off += (size_t)MTOK * DIM;                // 65,536
    float* hbuf     = ws + off; off += (size_t)MTOK * DIM;
    float* qb       = ws + off; off += (size_t)MTOK * INNER;
    float* kvb      = ws + off; off += (size_t)MTOK * 2 * INNER;          // 131,072
    float* attn_out = ws + off; off += (size_t)MTOK * INNER;
    float* ub       = ws + off; off += (size_t)MTOK * FF_WIDE;            // 524,288
    float* ffb      = ws + off; off += (size_t)MTOK * FF_HID;             // 262,144
    float* hiddens  = ws + off; off += (size_t)DEPTH * MTOK * DIM;        // 393,216
    float* aggb     = ws + off; off += (size_t)MTOK * DIM;
    float* mkvb     = ws + off; off += (size_t)MTOK * 2 * INNER;

    const float scale = 0.125f;  // DIM_HEAD^-0.5
    const int CE = MTOK * DIM;   // 65536 elems per chunk tensor

    // 1) embedding
    embed_kernel<<<BATCH * SEQLEN, 128, 0, stream>>>(tokens, tok_emb, x);

    // 2) chunk loop
    for (int c = 0; c < NCHUNK; ++c) {
        const int memLen = c * SEQ_CHUNK;
        gather_chunk<<<CE / 256, 256, 0, stream>>>(x, xc, c);

        for (int l = 0; l < DEPTH; ++l) {
            // keep pre-layer hidden for memory aggregation
            copy_f32<<<CE / 256, 256, 0, stream>>>(xc, hiddens + (size_t)l * CE, CE);

            // LN1
            ln_kernel<<<MTOK, 128, 0, stream>>>(xc, ln1_g + l * DIM, ln1_b + l * DIM, hbuf);

            // q = h @ Wq * scale ; kv = h @ Wkv
            {
                dim3 g(INNER / 128, MTOK / 16);
                gemm_bf16_wmma<<<g, 128, 0, stream>>>(hbuf, Wq + (size_t)l * DIM * INNER,
                                                      qb, nullptr, nullptr,
                                                      MTOK, INNER, DIM, scale);
            }
            {
                dim3 g(2 * INNER / 128, MTOK / 16);
                gemm_bf16_wmma<<<g, 128, 0, stream>>>(hbuf, Wkv + (size_t)l * DIM * 2 * INNER,
                                                      kvb, nullptr, nullptr,
                                                      MTOK, 2 * INNER, DIM, 1.f);
            }

            // attention (with growing memory)
            attn_kernel<<<BATCH * HEADS, 256, 0, stream>>>(qb, kvb, mem_k, mem_v,
                                                           memLen, rel_emb, attn_out);

            // xc = attn_out @ Wo + bo + xc
            {
                dim3 g(DIM / 128, MTOK / 16);
                gemm_bf16_wmma<<<g, 128, 0, stream>>>(attn_out, Wo + (size_t)l * INNER * DIM,
                                                      xc, bo + l * DIM, xc,
                                                      MTOK, DIM, INNER, 1.f);
            }

            // LN2 + GEGLU FFN
            ln_kernel<<<MTOK, 128, 0, stream>>>(xc, ln2_g + l * DIM, ln2_b + l * DIM, hbuf);
            {
                dim3 g(FF_WIDE / 128, MTOK / 16);
                gemm_bf16_wmma<<<g, 128, 0, stream>>>(hbuf, W1 + (size_t)l * DIM * FF_WIDE,
                                                      ub, b1 + l * FF_WIDE, nullptr,
                                                      MTOK, FF_WIDE, DIM, 1.f);
            }
            gelu_gate_kernel<<<(MTOK * FF_HID) / 256, 256, 0, stream>>>(ub, ffb);
            {
                dim3 g(DIM / 128, MTOK / 16);
                gemm_bf16_wmma<<<g, 128, 0, stream>>>(ffb, W2 + (size_t)l * FF_HID * DIM,
                                                      xc, b2 + l * DIM, xc,
                                                      MTOK, DIM, FF_HID, 1.f);
            }
        }

        scatter_chunk<<<CE / 256, 256, 0, stream>>>(xc, x, c);

        // memory update: agg = sum_l lw[l]*hidden_l ; [mk|mv] = agg @ W_mem_kv
        agg_kernel<<<CE / 256, 256, 0, stream>>>(hiddens, lw, aggb);
        {
            dim3 g(2 * INNER / 128, MTOK / 16);
            gemm_bf16_wmma<<<g, 128, 0, stream>>>(aggb, W_mem_kv, mkvb,
                                                  nullptr, nullptr,
                                                  MTOK, 2 * INNER, DIM, 1.f);
        }
        scatter_memkv<<<CE / 256, 256, 0, stream>>>(mkvb, mem_k, mem_v, c);
    }

    // 3) final LN + logits (2048 x 512 @ 512 x 32000; Wlog lives in L2)
    ln_kernel<<<BATCH * SEQLEN, 128, 0, stream>>>(x, lnf_g, lnf_b, xn);
    {
        dim3 g(VOCAB / 128, (BATCH * SEQLEN) / 16);
        gemm_bf16_wmma<<<g, 128, 0, stream>>>(xn, Wlog, logits, blog, nullptr,
                                              BATCH * SEQLEN, VOCAB, DIM, 1.f);
    }
}
